// EntityCandidateGenerator_26972394619662
// MI455X (gfx1250) — matrix operand order repaired and checked by reading, not compiled, not run on hardware
//
#include <hip/hip_runtime.h>

typedef __attribute__((ext_vector_type(16))) __bf16 v16bf;
typedef __attribute__((ext_vector_type(8)))  __bf16 v8bf;
typedef __attribute__((ext_vector_type(8)))  float  v8f;
typedef __attribute__((ext_vector_type(4)))  float  v4f;

#define D768 768
#define K2C  200
#define TOPKC 100
#define MT 16          // m-tiles per block in k_sims (M = 256 = MT*16)

__device__ __forceinline__ unsigned f2key(float f) {
    unsigned u = __float_as_uint(f);
    return (u & 0x80000000u) ? ~u : (u | 0x80000000u);
}

// ---------------------------------------------------------------- mention pooling
__global__ void k_mention_emb(const float* __restrict__ text,
                              const int* __restrict__ msent,
                              const int* __restrict__ mstart,
                              const int* __restrict__ mlen,
                              float* __restrict__ me, int S) {
    int m = blockIdx.x;
    int s0 = mstart[m], L = mlen[m];
    const float* base = text + ((size_t)msent[m] * S + s0) * D768;
    float inv = 1.0f / (float)(L + 1);
    for (int d = threadIdx.x; d < D768; d += blockDim.x) {
        float acc = 0.f;
        for (int t = 0; t <= L; ++t) acc += base[(size_t)t * D768 + d];
        me[(size_t)m * D768 + d] = acc * inv;
    }
}

// ------------------------------------------------ query (bf16) + m_part(+bs1) per mention
__global__ void k_mention_linear(const float* __restrict__ me,
                                 const float* __restrict__ Wret,
                                 const float* __restrict__ bret,
                                 const float* __restrict__ Ws1,
                                 const float* __restrict__ bs1,
                                 __bf16* __restrict__ qbf,
                                 float* __restrict__ mpart) {
    __shared__ float row[D768];
    int m = blockIdx.x;
    for (int d = threadIdx.x; d < D768; d += blockDim.x) row[d] = me[(size_t)m * D768 + d];
    __syncthreads();
    for (int j = threadIdx.x; j < D768; j += blockDim.x) {
        const float* wr = Wret + (size_t)j * D768;
        float q  = bret[j];
        float mp = bs1[j];
        for (int k = 0; k < D768; ++k) {
            q  += row[k] * wr[k];
            mp += row[k] * Ws1[(size_t)k * D768 + j];
        }
        qbf[(size_t)m * D768 + j]   = (__bf16)q;
        mpart[(size_t)m * D768 + j] = mp;
    }
}

// ------------------------------------------------ transpose+convert Ws1 lower half -> bf16 [e][k]
__global__ void k_ws1t(const float* __restrict__ Ws1, __bf16* __restrict__ w1t) {
    int idx = blockIdx.x * blockDim.x + threadIdx.x;
    if (idx < D768 * D768) {
        int e = idx / D768, k = idx % D768;
        w1t[idx] = (__bf16)Ws1[(size_t)(D768 + k) * D768 + e];
    }
}

// ---------------------------------------------------------------- sims = query @ kb^T (WMMA bf16)
// One block (8 waves) owns a 128-column kb slice and computes ALL 256 mentions against it:
// each B fragment is loaded from HBM exactly once; A (query bf16, 384KB) is L2-resident.
// Per wave: 16 independent accumulators (16 m-tiles x 16 n) -> hazard-free WMMA pipelining.
__global__ __launch_bounds__(256, 1) void k_sims(const __bf16* __restrict__ qbf,
                                                 const float* __restrict__ kb,
                                                 float* __restrict__ sims, int N) {
    int wave = threadIdx.x >> 5;
    int lane = threadIdx.x & 31;
    int half = lane >> 4, lr = lane & 15;
    int ntile = blockIdx.x * 128 + wave * 16;
    int nB = ntile + lr;
    bool nok = nB < N;
    const float*  brow  = kb + (size_t)(nok ? nB : 0) * D768;
    const __bf16* abase = qbf + (size_t)lr * D768;   // row lr of m-tile 0; tile t adds t*16 rows

    v8f acc[MT] = {};
    for (int k0 = 0; k0 < D768; k0 += 32) {
        // B: lane holds 16 consecutive K of column n=lr, K-half by lane/16 (f32 -> bf16 in-reg)
        const v4f* bp = (const v4f*)(brow + k0 + half * 16);
        v4f b0 = bp[0], b1 = bp[1], b2 = bp[2], b3 = bp[3];
        v16bf b;
#pragma unroll
        for (int i = 0; i < 4; ++i) {
            b[i]      = (__bf16)b0[i];
            b[4 + i]  = (__bf16)b1[i];
            b[8 + i]  = (__bf16)b2[i];
            b[12 + i] = (__bf16)b3[i];
        }
#pragma unroll
        for (int t = 0; t < MT; ++t) {
            // A: lane holds 16 consecutive K, K-half by lane/16 (matches B)
            const __bf16* arow = abase + (size_t)t * 16 * D768;
            v8bf c0 = *(const v8bf*)(arow + k0 + half * 16);
            v8bf c1 = *(const v8bf*)(arow + k0 + half * 16 + 8);
            v16bf a;
#pragma unroll
            for (int i = 0; i < 8; ++i) { a[i] = c0[i]; a[8 + i] = c1[i]; }
            acc[t] = __builtin_amdgcn_wmma_f32_16x16x32_bf16(false, a, false, b, (short)0, acc[t], false, false);
        }
    }
    if (!nok) return;  // divergence only after all WMMAs (EXEC all-ones during WMMA)
#pragma unroll
    for (int t = 0; t < MT; ++t)
#pragma unroll
        for (int r = 0; r < 8; ++r)
            sims[(size_t)(t * 16 + r + 8 * half) * (size_t)N + (size_t)nB] = acc[t][r];
}

// ---------------------------------------------------------------- radix-select top-K2 per mention
__global__ void k_select(const float* __restrict__ sims, int* __restrict__ cand, int N) {
    __shared__ unsigned hist[256];
    __shared__ unsigned s_prefix, s_need, cGt, cEq;
    int m = blockIdx.x;
    const float* row = sims + (size_t)m * N;
    if (threadIdx.x == 0) { s_prefix = 0u; s_need = (unsigned)K2C; }
    __syncthreads();

    for (int byte = 3; byte >= 0; --byte) {
        for (int i = threadIdx.x; i < 256; i += blockDim.x) hist[i] = 0u;
        __syncthreads();
        unsigned prefix = s_prefix;
        int shift = byte * 8;
        unsigned pmask = (byte == 3) ? 0u : (0xFFFFFFFFu << (shift + 8));
        for (int i = threadIdx.x; i < N; i += blockDim.x) {
            unsigned key = f2key(row[i]);
            if ((key & pmask) == prefix)
                atomicAdd(&hist[(key >> shift) & 0xFFu], 1u);
        }
        __syncthreads();
        if (threadIdx.x == 0) {
            unsigned need = s_need;
            unsigned acc = 0; int b = 255;
            while (true) { acc += hist[b]; if (acc >= need || b == 0) break; --b; }
            unsigned above = acc - hist[b];
            s_need = need - above;               // >=1 by construction
            s_prefix = prefix | ((unsigned)b << shift);
        }
        __syncthreads();
    }
    unsigned T = s_prefix;
    unsigned numGt = (unsigned)K2C - s_need;
    if (threadIdx.x == 0) { cGt = 0u; cEq = 0u; }
    __syncthreads();
    for (int i = threadIdx.x; i < N; i += blockDim.x) {
        unsigned key = f2key(row[i]);
        if (key > T) {
            unsigned p = atomicAdd(&cGt, 1u);
            if (p < numGt) cand[(size_t)m * K2C + p] = i;
        } else if (key == T) {
            unsigned p = numGt + atomicAdd(&cEq, 1u);
            if (p < (unsigned)K2C) cand[(size_t)m * K2C + p] = i;
        }
    }
}

// ---------------------------------------------------------------- candidate scorer (WMMA bf16, fused relu+dot+sigmoid)
// grid = (ceil(K2/16), M); block = 256 = 8 waves; wave covers e = p*128 + wave*16, p=0..5
__global__ void k_score(const float* __restrict__ kb,
                        const int* __restrict__ cand,
                        const __bf16* __restrict__ w1t,
                        const float* __restrict__ mpart,
                        const float* __restrict__ Ws2,
                        const float* __restrict__ bs2,
                        float* __restrict__ scores) {
    __shared__ float sc[16];
    int m = blockIdx.y;
    int ctile = blockIdx.x * 16;
    int wave = threadIdx.x >> 5;
    int lane = threadIdx.x & 31;
    int half = lane >> 4, lr = lane & 15;

    int cpos = ctile + lr;
    int ci = (cpos < K2C) ? cand[(size_t)m * K2C + cpos] : 0;
    const float* arow = kb + (size_t)ci * D768;

    if (threadIdx.x < 16) sc[threadIdx.x] = 0.f;
    __syncthreads();

    v8f acc[6] = {};
    for (int k0 = 0; k0 < D768; k0 += 32) {
        v4f a00 = *(const v4f*)(arow + k0 + half * 16);
        v4f a01 = *(const v4f*)(arow + k0 + half * 16 + 4);
        v4f a10 = *(const v4f*)(arow + k0 + half * 16 + 8);
        v4f a11 = *(const v4f*)(arow + k0 + half * 16 + 12);
        v16bf a;
#pragma unroll
        for (int i = 0; i < 4; ++i) {
            a[i]      = (__bf16)a00[i];
            a[4 + i]  = (__bf16)a01[i];
            a[8 + i]  = (__bf16)a10[i];
            a[12 + i] = (__bf16)a11[i];
        }
#pragma unroll
        for (int p = 0; p < 6; ++p) {
            int e0 = p * 128 + wave * 16 + lr;
            v16bf b = *(const v16bf*)(w1t + (size_t)e0 * D768 + k0 + half * 16);
            acc[p] = __builtin_amdgcn_wmma_f32_16x16x32_bf16(false, a, false, b, (short)0, acc[p], false, false);
        }
    }
    float part[8] = {0.f, 0.f, 0.f, 0.f, 0.f, 0.f, 0.f, 0.f};
#pragma unroll
    for (int p = 0; p < 6; ++p) {
        int e = p * 128 + wave * 16 + lr;
        float w2 = Ws2[e];
        float mp = mpart[(size_t)m * D768 + e];
#pragma unroll
        for (int r = 0; r < 8; ++r) {
            float h = acc[p][r] + mp;          // bs1 folded into mpart
            h = h > 0.f ? h : 0.f;
            part[r] += h * w2;
        }
    }
#pragma unroll
    for (int r = 0; r < 8; ++r) atomicAdd(&sc[r + 8 * half], part[r]);
    __syncthreads();
    if (threadIdx.x < 16 && ctile + (int)threadIdx.x < K2C) {
        float x = sc[threadIdx.x] + bs2[0];
        scores[(size_t)m * K2C + ctile + threadIdx.x] = 1.f / (1.f + __expf(-x));
    }
}

// ---------------------------------------------------------------- top-100 of 200: LDS bitonic sort
__global__ void k_topk(const float* __restrict__ scores,
                       const int* __restrict__ cand,
                       float* __restrict__ outs,
                       int* __restrict__ outi) {
    __shared__ float skey[256];
    __shared__ int   sidx[256];
    int m = blockIdx.x;
    int t = threadIdx.x;
    skey[t] = (t < K2C) ? scores[(size_t)m * K2C + t] : -3.4e38f;
    sidx[t] = (t < K2C) ? cand[(size_t)m * K2C + t] : 0;
    for (int size = 2; size <= 256; size <<= 1) {
        for (int stride = size >> 1; stride > 0; stride >>= 1) {
            __syncthreads();
            int p = t ^ stride;
            if (p > t) {
                bool descBlock = ((t & size) == 0);
                bool doSwap = descBlock ? (skey[t] < skey[p]) : (skey[t] > skey[p]);
                if (doSwap) {
                    float fk = skey[t]; skey[t] = skey[p]; skey[p] = fk;
                    int ii = sidx[t]; sidx[t] = sidx[p]; sidx[p] = ii;
                }
            }
        }
    }
    __syncthreads();
    if (t < TOPKC) {
        outs[(size_t)m * TOPKC + t] = skey[t];
        outi[(size_t)m * TOPKC + t] = sidx[t];
    }
}

// ----------------------------------------------------------------------------------------
extern "C" void kernel_launch(void* const* d_in, const int* in_sizes, int n_in,
                              void* d_out, int out_size, void* d_ws, size_t ws_size,
                              hipStream_t stream) {
    const float* text  = (const float*)d_in[0];
    const float* kb    = (const float*)d_in[1];
    const float* Wret  = (const float*)d_in[2];
    const float* bret  = (const float*)d_in[3];
    const float* Ws1   = (const float*)d_in[4];
    const float* bs1   = (const float*)d_in[5];
    const float* Ws2   = (const float*)d_in[6];
    const float* bs2   = (const float*)d_in[7];
    const int* msent   = (const int*)d_in[8];
    const int* mstart  = (const int*)d_in[9];
    const int* mlen    = (const int*)d_in[10];

    const int M = in_sizes[8];                 // 256
    const int N = in_sizes[1] / D768;          // 200000

    // workspace carve-up
    char* w = (char*)d_ws;
    float*  me     = (float*)w;   w += (size_t)M * D768 * sizeof(float);
    float*  mpart  = (float*)w;   w += (size_t)M * D768 * sizeof(float);
    __bf16* qbf    = (__bf16*)w;  w += (size_t)M * D768 * sizeof(__bf16);
    __bf16* w1t    = (__bf16*)w;  w += (size_t)D768 * D768 * sizeof(__bf16);
    int*    cand   = (int*)w;     w += (size_t)M * K2C * sizeof(int);
    float*  scr    = (float*)w;   w += (size_t)M * K2C * sizeof(float);
    float*  sims   = (float*)w;   // M * N floats (~205 MB)

    float* outs = (float*)d_out;
    int*   outi = (int*)((float*)d_out + (size_t)M * TOPKC);

    k_mention_emb<<<M, 256, 0, stream>>>(text, msent, mstart, mlen, me, 512);
    k_mention_linear<<<M, 256, 0, stream>>>(me, Wret, bret, Ws1, bs1, qbf, mpart);
    k_ws1t<<<(D768 * D768 + 255) / 256, 256, 0, stream>>>(Ws1, w1t);
    k_sims<<<dim3((N + 127) / 128), 256, 0, stream>>>(qbf, kb, sims, N);
    k_select<<<M, 256, 0, stream>>>(sims, cand, N);
    k_score<<<dim3((K2C + 15) / 16, M), 256, 0, stream>>>(kb, cand, w1t, mpart, Ws2, bs2, scr);
    k_topk<<<M, 256, 0, stream>>>(scr, cand, outs, outi);
}
